// StackGNN_71794673320502
// MI455X (gfx1250) — compile-verified, matrix operand order checked by reading, not run on hardware
//
#include <hip/hip_runtime.h>

typedef __attribute__((ext_vector_type(16))) _Float16 v16h;
typedef __attribute__((ext_vector_type(8)))  _Float16 v8h;
typedef __attribute__((ext_vector_type(8)))  float    v8f;

// ---------------------------------------------------------------------------
// Fragment helpers
// ---------------------------------------------------------------------------
template<int BASE>
__device__ __forceinline__ void fragF32(const float* __restrict__ p, int off, v16h& a) {
  const float4 q0 = *(const float4*)(p + off);
  const float4 q1 = *(const float4*)(p + off + 4);
  a[BASE+0] = (_Float16)q0.x; a[BASE+1] = (_Float16)q0.y;
  a[BASE+2] = (_Float16)q0.z; a[BASE+3] = (_Float16)q0.w;
  a[BASE+4] = (_Float16)q1.x; a[BASE+5] = (_Float16)q1.y;
  a[BASE+6] = (_Float16)q1.z; a[BASE+7] = (_Float16)q1.w;
}

template<int BASE>
__device__ __forceinline__ void fragF16(const _Float16* p, int off, v16h& a) {
  const v8h q = *(const v8h*)(p + off);
#pragma unroll
  for (int j = 0; j < 8; ++j) a[BASE + j] = q[j];
}

__device__ __forceinline__ void atomicMaxF(float* addr, float val) {
  unsigned int* ua = (unsigned int*)addr;
  unsigned int old = *ua;
  while (__uint_as_float(old) < val) {
    unsigned int assumed = old;
    old = atomicCAS(ua, assumed, __float_as_uint(val));
    if (old == assumed) break;
  }
}

// ---------------------------------------------------------------------------
// Repack W[K,Nn] (row-major f32) into per-lane WMMA B fragments (f16).
// Fragment index: ((ntile*(K/32) + kstep)*32 + lane) * 16 halves.
// ---------------------------------------------------------------------------
__global__ void pack_w_k(const float* __restrict__ W, _Float16* __restrict__ out,
                         int K, int Nn) {
  int idx = blockIdx.x * 64 + threadIdx.x;
  int total = (Nn >> 4) * (K >> 5) * 32;
  if (idx >= total) return;
  int lane  = idx & 31;
  int t     = idx >> 5;
  int ks    = K >> 5;
  int kstep = t % ks;
  int ntile = t / ks;
  int col   = ntile * 16 + (lane & 15);
  int k0    = kstep * 32 + ((lane >> 4) << 3);
  _Float16* o = out + (size_t)idx * 16;
#pragma unroll
  for (int j = 0; j < 8; ++j) {
    o[j]     = (_Float16)W[(size_t)(k0 + j)      * Nn + col];
    o[j + 8] = (_Float16)W[(size_t)(k0 + 16 + j) * Nn + col];
  }
}

// ---------------------------------------------------------------------------
// Encoder GEMM: out(f16) = relu(A_f32[M,K] @ W[K,Nn] + bias).
// blockDim = 32*(Nn/16); grid = ceil(M/16).
// ---------------------------------------------------------------------------
__global__ void gemm_enc_k(const float* __restrict__ A,
                           const _Float16* __restrict__ Wp,
                           const float* __restrict__ bias,
                           _Float16* __restrict__ out,
                           int M, int K, int Nn) {
  const int lane  = threadIdx.x & 31;
  const int ntile = threadIdx.x >> 5;
  const int col   = ntile * 16 + (lane & 15);
  int m = blockIdx.x * 16 + (lane & 15);
  if (m >= M) m = M - 1;
  const float* Ar  = A + (size_t)m * K;
  const int khi    = (lane >> 4) << 3;
  const int ksteps = K >> 5;
  const v16h* Bf   = (const v16h*)Wp;
  v8f c = {};
  for (int kk = 0; kk < K; kk += 32) {
    const int k0 = kk + khi;
    v16h a;
    fragF32<0>(Ar, k0, a);
    fragF32<8>(Ar, k0 + 16, a);
    v16h b = Bf[((size_t)ntile * ksteps + (kk >> 5)) * 32 + lane];
    c = __builtin_amdgcn_wmma_f32_16x16x32_f16(false, a, false, b,
                                               (short)0, c, false, false);
  }
  const float bv    = bias[col];
  const int   rbase = blockIdx.x * 16 + ((lane >> 4) << 3);
#pragma unroll
  for (int r = 0; r < 8; ++r) {
    int gm = rbase + r;
    if (gm < M)
      out[(size_t)gm * Nn + col] = (_Float16)fmaxf(c[r] + bv, 0.0f);
  }
}

// ---------------------------------------------------------------------------
// Head GEMM: out(f32) = A_f16[M,128] @ W[128,Nn] + bias   (no activation)
// ---------------------------------------------------------------------------
__global__ void gemm_head_k(const _Float16* __restrict__ A,
                            const _Float16* __restrict__ Wp,
                            const float* __restrict__ bias,
                            float* __restrict__ out,
                            int M, int Nn) {
  const int lane  = threadIdx.x & 31;
  const int ntile = threadIdx.x >> 5;
  const int col   = ntile * 16 + (lane & 15);
  int m = blockIdx.x * 16 + (lane & 15);
  if (m >= M) m = M - 1;
  const _Float16* Ar = A + (size_t)m * 128;
  const int khi = (lane >> 4) << 3;
  const v16h* Bf = (const v16h*)Wp;
  v8f c = {};
  for (int kk = 0; kk < 128; kk += 32) {
    const int k0 = kk + khi;
    v16h a;
    fragF16<0>(Ar, k0, a);
    fragF16<8>(Ar, k0 + 16, a);
    v16h b = Bf[((size_t)ntile * 4 + (kk >> 5)) * 32 + lane];
    c = __builtin_amdgcn_wmma_f32_16x16x32_f16(false, a, false, b,
                                               (short)0, c, false, false);
  }
  const float bv    = bias[col];
  const int   rbase = blockIdx.x * 16 + ((lane >> 4) << 3);
#pragma unroll
  for (int r = 0; r < 8; ++r) {
    int gm = rbase + r;
    if (gm < M)
      out[(size_t)gm * Nn + col] = c[r] + bv;
  }
}

// ---------------------------------------------------------------------------
// Edge update: e_out = relu(concat(h[src], h[dst], e_in) @ W[384,128] + b).
// A tile (16 rows x 384 halves) staged once per block into LDS using the
// gfx1250 async global->LDS path (GLOBAL_LOAD_ASYNC_TO_LDS_B128, ASYNCcnt),
// then all 8 waves build fragments via ds_load_b128. Row stride padded to
// 392 halves so the 16 row-lanes hit disjoint bank groups.
// blockDim = 256, grid = ceil(E/16).
// ---------------------------------------------------------------------------
#define ETS 392   // padded LDS row stride in halves

__global__ void edge_update_k(const _Float16* __restrict__ h,
                              const _Float16* __restrict__ ein,
                              const int* __restrict__ src,
                              const int* __restrict__ dst,
                              const _Float16* __restrict__ Wp,
                              const float* __restrict__ bias,
                              _Float16* __restrict__ eout, int E_) {
  __shared__ _Float16 tile[16 * ETS];

  // ---- async cooperative stage: 768 chunks of 8 halves (16B) ----
  for (int c = threadIdx.x; c < 16 * 48; c += 256) {
    const int row = c / 48;
    const int k   = (c % 48) * 8;
    int i = blockIdx.x * 16 + row;
    if (i >= E_) i = E_ - 1;
    const _Float16* p;
    if (k < 128)       p = h   + (size_t)src[i] * 128 + k;
    else if (k < 256)  p = h   + (size_t)dst[i] * 128 + (k - 128);
    else               p = ein + (size_t)i      * 128 + (k - 256);
    // LDS byte offset = low 32 bits of the generic shared-memory pointer.
    unsigned int       lds_off = (unsigned int)(size_t)(&tile[row * ETS + k]);
    unsigned long long gaddr   = (unsigned long long)(size_t)p;
    asm volatile("global_load_async_to_lds_b128 %0, %1, off"
                 :: "v"(lds_off), "v"(gaddr) : "memory");
  }
  asm volatile("s_wait_asynccnt 0x0" ::: "memory");
  __syncthreads();

  // ---- compute ----
  const int lane  = threadIdx.x & 31;
  const int ntile = threadIdx.x >> 5;
  const int col   = ntile * 16 + (lane & 15);
  const int r     = lane & 15;
  const int khi   = (lane >> 4) << 3;
  const v16h* Bf  = (const v16h*)Wp;
  v8f c = {};
#pragma unroll
  for (int kk = 0; kk < 12; ++kk) {
    const int k0 = kk * 32 + khi;
    v16h a;
    fragF16<0>(&tile[r * ETS], k0, a);
    fragF16<8>(&tile[r * ETS], k0 + 16, a);
    v16h b = Bf[((size_t)ntile * 12 + kk) * 32 + lane];
    c = __builtin_amdgcn_wmma_f32_16x16x32_f16(false, a, false, b,
                                               (short)0, c, false, false);
  }
  const float bv    = bias[col];
  const int   rbase = blockIdx.x * 16 + ((lane >> 4) << 3);
#pragma unroll
  for (int rr = 0; rr < 8; ++rr) {
    int gm = rbase + rr;
    if (gm < E_)
      eout[(size_t)gm * 128 + col] = (_Float16)fmaxf(c[rr] + bv, 0.0f);
  }
}

// ---------------------------------------------------------------------------
// Node update: h_out(f16) = relu(concat(h_f16, agg_f32) @ W[256,128] + b).
// ---------------------------------------------------------------------------
__global__ void node_update_k(const _Float16* __restrict__ h,
                              const float* __restrict__ agg,
                              const _Float16* __restrict__ Wp,
                              const float* __restrict__ bias,
                              _Float16* __restrict__ hout, int N_) {
  const int lane  = threadIdx.x & 31;
  const int ntile = threadIdx.x >> 5;
  const int col   = ntile * 16 + (lane & 15);
  int m = blockIdx.x * 16 + (lane & 15);
  if (m >= N_) m = N_ - 1;
  const _Float16* hr = h   + (size_t)m * 128;
  const float*    ar = agg + (size_t)m * 128;
  const int khi = (lane >> 4) << 3;
  const v16h* Bf = (const v16h*)Wp;
  v8f c = {};
#pragma unroll
  for (int kk = 0; kk < 8; ++kk) {
    const int k0 = kk * 32 + khi;
    v16h a;
    if (k0 < 128) fragF16<0>(hr, k0, a);
    else          fragF32<0>(ar, k0 - 128, a);
    if (k0 + 16 < 128) fragF16<8>(hr, k0 + 16, a);
    else               fragF32<8>(ar, k0 + 16 - 128, a);
    v16h b = Bf[((size_t)ntile * 8 + kk) * 32 + lane];
    c = __builtin_amdgcn_wmma_f32_16x16x32_f16(false, a, false, b,
                                               (short)0, c, false, false);
  }
  const float bv    = bias[col];
  const int   rbase = blockIdx.x * 16 + ((lane >> 4) << 3);
#pragma unroll
  for (int r = 0; r < 8; ++r) {
    int gm = rbase + r;
    if (gm < N_)
      hout[(size_t)gm * 128 + col] = (_Float16)fmaxf(c[r] + bv, 0.0f);
  }
}

// ---------------------------------------------------------------------------
// Attention
// ---------------------------------------------------------------------------
__global__ void logit_max_k(const _Float16* __restrict__ e,
                            const float* __restrict__ Wa,
                            const float* __restrict__ ba,
                            const int* __restrict__ dst,
                            float* __restrict__ logit,
                            float* __restrict__ maxb, int E_) {
  const int lane = threadIdx.x & 31;
  const int wid  = threadIdx.x >> 5;
  const int i    = blockIdx.x * 8 + wid;
  if (i >= E_) return;
  const _Float16* er = e + (size_t)i * 128;
  float s = 0.0f;
#pragma unroll
  for (int j = 0; j < 4; ++j) {
    int f = lane + j * 32;
    s += (float)er[f] * Wa[f];
  }
#pragma unroll
  for (int m = 16; m >= 1; m >>= 1) s += __shfl_xor(s, m, 32);
  if (lane == 0) {
    float lg = s + ba[0];
    logit[i] = lg;
    atomicMaxF(&maxb[dst[i]], lg);
  }
}

__global__ void exp_sum_k(float* __restrict__ logit,
                          const float* __restrict__ maxb,
                          float* __restrict__ denom,
                          const int* __restrict__ dst, int E_) {
  int i = blockIdx.x * 256 + threadIdx.x;
  if (i >= E_) return;
  int d   = dst[i];
  float w = __expf(logit[i] - maxb[d]);
  logit[i] = w;
  atomicAdd(&denom[d], w);
}

__global__ void agg_k(const _Float16* __restrict__ e,
                      const float* __restrict__ logit,
                      const float* __restrict__ denom,
                      const int* __restrict__ dst,
                      float* __restrict__ agg, int E_) {
  const int lane = threadIdx.x & 31;
  const int wid  = threadIdx.x >> 5;
  const int i    = blockIdx.x * 8 + wid;
  if (i >= E_) return;
  const int   d     = dst[i];
  const float coef  = logit[i] / (denom[d] + 1e-9f);
  const _Float16* er = e + (size_t)i * 128;
  float*          ar = agg + (size_t)d * 128;
#pragma unroll
  for (int j = 0; j < 4; ++j) {
    int f = lane + j * 32;
    atomicAdd(&ar[f], coef * (float)er[f]);
  }
}

__global__ void init_attn_k(float* __restrict__ maxb, float* __restrict__ denom,
                            float* __restrict__ agg, int N_) {
  int i = blockIdx.x * 256 + threadIdx.x;
  if (i < N_) { maxb[i] = -__builtin_huge_valf(); denom[i] = 0.0f; }
  if (i < N_ * 128) agg[i] = 0.0f;
}

// ---------------------------------------------------------------------------
extern "C" void kernel_launch(void* const* d_in, const int* in_sizes, int n_in,
                              void* d_out, int out_size, void* d_ws, size_t ws_size,
                              hipStream_t stream) {
  const float* nf     = (const float*)d_in[0];
  const float* ef     = (const float*)d_in[1];
  const int*   src    = (const int*)d_in[2];
  const int*   dst    = (const int*)d_in[3];
  const float* Wn_enc = (const float*)d_in[4];
  const float* bn_enc = (const float*)d_in[5];
  const float* We_enc = (const float*)d_in[6];
  const float* be_enc = (const float*)d_in[7];
  const float* We_gn  = (const float*)d_in[8];
  const float* be_gn  = (const float*)d_in[9];
  const float* Wa_gn  = (const float*)d_in[10];
  const float* ba_gn  = (const float*)d_in[11];
  const float* Wn_gn  = (const float*)d_in[12];
  const float* bn_gn  = (const float*)d_in[13];
  const float* Wno    = (const float*)d_in[14];
  const float* bno    = (const float*)d_in[15];
  const float* Weo    = (const float*)d_in[16];
  const float* beo    = (const float*)d_in[17];

  const int N = in_sizes[0] / 64;   // 50000
  const int E = in_sizes[1] / 32;   // 800000
  const int L = 3;

  const size_t NH = (size_t)N * 128;
  const size_t EH = (size_t)E * 128;

  // f32 region first, then f16 region (all sub-sizes keep 16B alignment)
  float* logit = (float*)d_ws;
  float* maxb  = logit + E;
  float* denom = maxb + N;
  float* agg   = denom + N;
  _Float16* h  = (_Float16*)(agg + NH);
  _Float16* h2 = h  + NH;
  _Float16* eA = h2 + NH;
  _Float16* eB = eA + EH;
  _Float16* wpk = eB + EH;

  _Float16* pWn_enc = wpk;                              // 64*128
  _Float16* pWe_enc = pWn_enc + 64 * 128;               // 32*128
  _Float16* pWe_gn  = pWe_enc + 32 * 128;               // L*384*128
  _Float16* pWn_gn  = pWe_gn + (size_t)L * 384 * 128;   // L*256*128
  _Float16* pWno    = pWn_gn + (size_t)L * 256 * 128;   // 128*64
  _Float16* pWeo    = pWno + 128 * 64;                  // 128*32

  auto packGrid = [](int K, int Nn) { return ((Nn >> 4) * (K >> 5) * 32 + 63) / 64; };

  pack_w_k<<<packGrid(64, 128), 64, 0, stream>>>(Wn_enc, pWn_enc, 64, 128);
  pack_w_k<<<packGrid(32, 128), 64, 0, stream>>>(We_enc, pWe_enc, 32, 128);
  for (int l = 0; l < L; ++l) {
    pack_w_k<<<packGrid(384, 128), 64, 0, stream>>>(
        We_gn + (size_t)l * 384 * 128, pWe_gn + (size_t)l * 384 * 128, 384, 128);
    pack_w_k<<<packGrid(256, 128), 64, 0, stream>>>(
        Wn_gn + (size_t)l * 256 * 128, pWn_gn + (size_t)l * 256 * 128, 256, 128);
  }
  pack_w_k<<<packGrid(128, 64), 64, 0, stream>>>(Wno, pWno, 128, 64);
  pack_w_k<<<packGrid(128, 32), 64, 0, stream>>>(Weo, pWeo, 128, 32);

  const int gN = (N + 15) / 16;
  const int gE = (E + 15) / 16;

  // Encoders (f32 in -> f16 hidden)
  gemm_enc_k<<<gN, 256, 0, stream>>>(nf, pWn_enc, bn_enc, h,  N, 64, 128);
  gemm_enc_k<<<gE, 256, 0, stream>>>(ef, pWe_enc, be_enc, eA, E, 32, 128);

  _Float16* hc = h, *hn = h2, *ec = eA, *en = eB;
  for (int l = 0; l < L; ++l) {
    edge_update_k<<<gE, 256, 0, stream>>>(hc, ec, src, dst,
        pWe_gn + (size_t)l * 384 * 128, be_gn + (size_t)l * 128, en, E);

    init_attn_k<<<(N * 128 + 255) / 256, 256, 0, stream>>>(maxb, denom, agg, N);
    logit_max_k<<<(E + 7) / 8, 256, 0, stream>>>(en, Wa_gn + (size_t)l * 128,
                                                 ba_gn + l, dst, logit, maxb, E);
    exp_sum_k<<<(E + 255) / 256, 256, 0, stream>>>(logit, maxb, denom, dst, E);
    agg_k<<<(E + 7) / 8, 256, 0, stream>>>(en, logit, denom, dst, agg, E);

    node_update_k<<<gN, 256, 0, stream>>>(hc, agg,
        pWn_gn + (size_t)l * 256 * 128, bn_gn + (size_t)l * 128, hn, N);

    _Float16* t;
    t = hc; hc = hn; hn = t;
    t = ec; ec = en; en = t;
  }

  // Output heads (f16 hidden -> f32 out)
  float* out = (float*)d_out;
  gemm_head_k<<<gN, 128, 0, stream>>>(hc, pWno, bno, out, N, 64);
  gemm_head_k<<<gE, 64, 0, stream>>>(ec, pWeo, beo, out + (size_t)N * 64, E, 32);
}
